// MergeEncoder_47768626266491
// MI455X (gfx1250) — matrix operand order repaired and checked by reading, not compiled, not run on hardware
//
#include <hip/hip_runtime.h>

// ---------------------------------------------------------------------------
// GIN on a fully-connected graph (N=1024, F=128, H=256).
//   * dense-edge segment_sum == column-sum broadcast to rows 1..N-1
//   * GEMMs: v_wmma_f32_16x16x32_bf16 (bf16 in, f32 accum), 128x128 tiles
//   * tile staging: TENSOR_LOAD_TO_LDS (TDM) with hardware LDS padding,
//     double-buffered, overlapped with WMMA via TENSORcnt
//   * inner loop: all 9 fragments loaded before the 8 WMMAs (no full-dscnt
//     stall per matrix op)
//   * all reductions: 2-stage deterministic trees (no float atomics)
// ---------------------------------------------------------------------------

typedef __attribute__((ext_vector_type(16))) __bf16        v16bf;
typedef __attribute__((ext_vector_type(8)))  float         v8f;
typedef __attribute__((ext_vector_type(4)))  unsigned int  u32x4;
typedef __attribute__((ext_vector_type(4)))  int           i32x4;
typedef __attribute__((ext_vector_type(8)))  int           i32x8;

#define BN_EPS 1e-5f
#define RB 16            // row-blocks for 2-stage reductions (1024/16 = 64 rows)

__device__ __forceinline__ unsigned short f32_to_bf16(float f) {
  union { float f; unsigned u; } v; v.f = f;
  unsigned r = v.u + 0x7FFFu + ((v.u >> 16) & 1u);   // round-to-nearest-even
  return (unsigned short)(r >> 16);
}

// ---------------- weights: f32 [K,Nc] -> bf16 transposed [Nc,K] ----------------
__global__ void cvt_bf16_t_kernel(const float* __restrict__ W,
                                  unsigned short* __restrict__ Bt, int K, int Nc) {
  int idx = blockIdx.x * blockDim.x + threadIdx.x;
  if (idx >= K * Nc) return;
  int k = idx / Nc, n = idx - k * Nc;
  Bt[(size_t)n * K + k] = f32_to_bf16(W[idx]);
}

// ---------------- 2-stage column sum ----------------
__global__ void colsum_part(const float* __restrict__ x, float* __restrict__ part,
                            int n, int c) {
  int col = blockIdx.x * blockDim.x + threadIdx.x;
  int rb  = blockIdx.y;
  if (col >= c) return;
  int rows = n / RB, r0 = rb * rows;
  float s = 0.f;
  for (int i = r0; i < r0 + rows; ++i) s += x[(size_t)i * c + col];
  part[rb * c + col] = s;
}
__global__ void colsum_fin(const float* __restrict__ part, float* __restrict__ S, int c) {
  int col = blockIdx.x * blockDim.x + threadIdx.x;
  if (col >= c) return;
  float s = 0.f;
  for (int rb = 0; rb < RB; ++rb) s += part[rb * c + col];
  S[col] = s;
}

// ---------------- GIN aggregate + bf16 convert ----------------
__global__ void agg_bf16_kernel(const float* __restrict__ x, const float* __restrict__ S,
                                unsigned short* __restrict__ out, int n, int c) {
  int idx = blockIdx.x * blockDim.x + threadIdx.x;
  if (idx >= n * c) return;
  int row = idx / c, col = idx - row * c;
  float v = x[idx] + (row > 0 ? S[col] : 0.f);
  out[idx] = f32_to_bf16(v);
}

// ---------------- 2-stage BN stats over relu(h) (biased var) ----------------
__global__ void bnstats_part(const float* __restrict__ h, float* __restrict__ part,
                             int n, int c) {
  int col = blockIdx.x * blockDim.x + threadIdx.x;
  int rb  = blockIdx.y;
  if (col >= c) return;
  int rows = n / RB, r0 = rb * rows;
  float s = 0.f, s2 = 0.f;
  for (int i = r0; i < r0 + rows; ++i) {
    float r = fmaxf(h[(size_t)i * c + col], 0.f);
    s += r; s2 += r * r;
  }
  part[rb * c + col]        = s;
  part[(RB + rb) * c + col] = s2;
}
__global__ void bnstats_fin(const float* __restrict__ part, float* __restrict__ stats,
                            int n, int c) {
  int col = blockIdx.x * blockDim.x + threadIdx.x;
  if (col >= c) return;
  float s = 0.f, s2 = 0.f;
  for (int rb = 0; rb < RB; ++rb) { s += part[rb * c + col]; s2 += part[(RB + rb) * c + col]; }
  float mu  = s / (float)n;
  stats[col]     = mu;
  stats[c + col] = s2 / (float)n - mu * mu;
}

// ---------------- BN apply on relu(h) -> f32 ----------------
__global__ void bn_apply_kernel(const float* __restrict__ h, const float* __restrict__ stats,
                                const float* __restrict__ gamma, const float* __restrict__ beta,
                                float* __restrict__ out, int n, int c) {
  int idx = blockIdx.x * blockDim.x + threadIdx.x;
  if (idx >= n * c) return;
  int col = idx % c;
  float r = fmaxf(h[idx], 0.f);
  float k = rsqrtf(stats[c + col] + BN_EPS) * gamma[col];
  out[idx] = (r - stats[col]) * k + beta[col];
}

// ---------------- final readout: 2-stage sum_i BN(relu(h))[i,c] ----------------
__global__ void bnred_part(const float* __restrict__ h, const float* __restrict__ stats,
                           const float* __restrict__ gamma, const float* __restrict__ beta,
                           float* __restrict__ part, int n, int c) {
  int col = blockIdx.x * blockDim.x + threadIdx.x;
  int rb  = blockIdx.y;
  if (col >= c) return;
  float mu = stats[col];
  float k  = rsqrtf(stats[c + col] + BN_EPS) * gamma[col];
  float be = beta[col];
  int rows = n / RB, r0 = rb * rows;
  float acc = 0.f;
  for (int i = r0; i < r0 + rows; ++i) {
    float r = fmaxf(h[(size_t)i * c + col], 0.f);
    acc += (r - mu) * k + be;
  }
  part[rb * c + col] = acc;
}
__global__ void bnred_fin(const float* __restrict__ part, float* __restrict__ out, int c) {
  int col = blockIdx.x * blockDim.x + threadIdx.x;
  if (col >= c) return;
  float s = 0.f;
  for (int rb = 0; rb < RB; ++rb) s += part[rb * c + col];
  out[col] = s;
}

// ---------------------------------------------------------------------------
// TDM: DMA one 128-row x 32-col bf16 tile (row stride = row_len elems) from
// global into LDS, padding each 64B row with 16B -> LDS row stride 80B (40
// ushorts), matching the aligned b128 fragment-read layout.
// D# per CDNA5 ISA:
//   g0: [1:0]=count=1, [63:32]=lds_addr, [120:64]=global_addr, [127:126]=2
//   g1: [17:16]=data_size=1(2B), [20]=pad_enable, [24:22]=pad_interval=3(16 DW),
//       [31:25]=pad_amount=3(4 DW), dim0=row_len, dim1=rows_total,
//       tile_dim0=32, tile_dim1=128, dim0_stride=row_len
//   g2/g3: zero (2-D tile; tile_dim2/3/4 = 0)
// ---------------------------------------------------------------------------
__device__ __forceinline__ void tdm_load_tile32(unsigned lds_off,
                                                const unsigned short* gsrc,
                                                unsigned row_len, unsigned rows_total) {
  unsigned long long ga = (unsigned long long)(size_t)gsrc;
  unsigned ga_lo = (unsigned)__builtin_amdgcn_readfirstlane((int)(unsigned)ga);
  unsigned ga_hi = (unsigned)__builtin_amdgcn_readfirstlane((int)(unsigned)(ga >> 32));
  unsigned ldo   = (unsigned)__builtin_amdgcn_readfirstlane((int)lds_off);
  unsigned rl    = (unsigned)__builtin_amdgcn_readfirstlane((int)row_len);
  unsigned rt    = (unsigned)__builtin_amdgcn_readfirstlane((int)rows_total);

  u32x4 g0 = { 1u,                                   // count=1, user descriptor
               ldo,                                  // lds_addr
               ga_lo,                                // global_addr[31:0]
               (ga_hi & 0x01FFFFFFu) | 0x80000000u };// global_addr[56:32] | type=2
  i32x8 g1 = { (int)((1u << 16) | (1u << 20) | (3u << 22) | (3u << 25)),
               (int)(rl << 16),                      // tensor_dim0[15:0]
               (int)((rl >> 16) | (rt << 16)),       // dim0 hi | tensor_dim1 lo
               (int)((rt >> 16) | (32u << 16)),      // dim1 hi | tile_dim0=32
               128,                                  // tile_dim1=128, tile_dim2=0
               (int)rl,                              // tensor_dim0_stride[31:0]
               0, 0 };
  i32x4 gz = { 0, 0, 0, 0 };
  asm volatile("tensor_load_to_lds %0, %1, %2, %3"
               :: "s"(g0), "s"(g1), "s"(gz), "s"(gz)
               : "memory");
}

// ---------------------------------------------------------------------------
// WMMA GEMM: C[M,NC] = A[M,K](bf16) @ Bt[NC,K](bf16)^T + bias.
// Block = 256 threads (8 wave32), 128x128 tile, K-step 32, double-buffered
// TDM tile staging.  Wave w owns rows [w*16,w*16+16) x all 128 cols.
// All 9 fragments are loaded before the 8 WMMAs so the matrix pipe runs
// back-to-back instead of stalling on a full s_wait_dscnt per tile.
// ---------------------------------------------------------------------------
#define TM 128
#define TN 128
#define LDS_S 40   // LDS row stride in ushorts: 32 data + 8 pad (TDM pad), 80B rows

template <int RELU, int OUT_BF16>
__global__ __launch_bounds__(256)
void gemm_bf16_wmma(const unsigned short* __restrict__ A,    // [M,K] row-major
                    const unsigned short* __restrict__ Bt,   // [NC,K] row-major
                    const float* __restrict__ bias,          // [NC]
                    void* __restrict__ Cout,                 // [M,NC] f32 or bf16
                    int M, int K, int NC) {
  __shared__ __align__(16) unsigned short ldsA[2][TM * LDS_S];
  __shared__ __align__(16) unsigned short ldsB[2][TN * LDS_S];

  const int tid  = threadIdx.x;
  const int wave = tid >> 5;
  const int lane = tid & 31;
  const int kg   = lane >> 4;   // K-group half of wave
  const int ln   = lane & 15;   // row (A) / col (B,C) within 16x16 tile

  const int rowBase = blockIdx.y * TM;
  const int colBase = blockIdx.x * TN;
  const int m0      = wave * 16;

  v8f acc[8];
  v8f zero = {0.f, 0.f, 0.f, 0.f, 0.f, 0.f, 0.f, 0.f};
#pragma unroll
  for (int t = 0; t < 8; ++t) acc[t] = zero;

  // prologue: stage 0 in-flight
  if (wave == 0) {
    tdm_load_tile32((unsigned)(size_t)&ldsA[0][0], A  + (size_t)rowBase * K, K, M);
    tdm_load_tile32((unsigned)(size_t)&ldsB[0][0], Bt + (size_t)colBase * K, K, NC);
  }

  int buf = 0;
  for (int k0 = 0; k0 < K; k0 += 32, buf ^= 1) {
    if (wave == 0) {
      if (k0 + 32 < K) {   // prefetch next stage, keep one stage in flight
        tdm_load_tile32((unsigned)(size_t)&ldsA[buf ^ 1][0],
                        A  + (size_t)rowBase * K + (k0 + 32), K, M);
        tdm_load_tile32((unsigned)(size_t)&ldsB[buf ^ 1][0],
                        Bt + (size_t)colBase * K + (k0 + 32), K, NC);
        __builtin_amdgcn_s_wait_tensorcnt((short)2);   // stage k0 complete
      } else {
        __builtin_amdgcn_s_wait_tensorcnt((short)0);
      }
    }
    __syncthreads();   // LDS stage `buf` ready for all waves

    // Load A fragment + all 8 B fragments first, then issue the 8 WMMAs.
    union Frag { v16bf v; u32x4 q[2]; };
    Frag afrag, bfrag[8];
    {
      const u32x4* p = (const u32x4*)(&ldsA[buf][(m0 + ln) * LDS_S + kg * 16]);
      afrag.q[0] = p[0];   // K = kg*16 + {0..7}
      afrag.q[1] = p[1];   // K = kg*16 + 8 + {0..7}   (contiguous, matches B)
    }
#pragma unroll
    for (int t = 0; t < 8; ++t) {
      // B fragment: col t*16+ln, K = kg*16+e (contiguous, same as A) == row of Bt
      const u32x4* p = (const u32x4*)(&ldsB[buf][(t * 16 + ln) * LDS_S + kg * 16]);
      bfrag[t].q[0] = p[0];
      bfrag[t].q[1] = p[1];
    }
#pragma unroll
    for (int t = 0; t < 8; ++t) {
      acc[t] = __builtin_amdgcn_wmma_f32_16x16x32_bf16(
          false, afrag.v, false, bfrag[t].v, (short)0, acc[t], false, false);
    }
    __syncthreads();   // done reading `buf` before TDM overwrites it next+1 iter
  }

  // epilogue: C element j of lane -> row = m0 + kg*8 + j, col = t*16 + ln
  const int outRow0 = rowBase + m0 + kg * 8;
#pragma unroll
  for (int t = 0; t < 8; ++t) {
    int col = colBase + t * 16 + ln;
    float b = bias[col];
#pragma unroll
    for (int j = 0; j < 8; ++j) {
      float v = acc[t][j] + b;
      if (RELU) v = fmaxf(v, 0.f);
      size_t o = (size_t)(outRow0 + j) * NC + col;
      if (OUT_BF16) ((unsigned short*)Cout)[o] = f32_to_bf16(v);
      else          ((float*)Cout)[o] = v;
    }
  }
}

// ---------------------------------------------------------------------------
extern "C" void kernel_launch(void* const* d_in, const int* in_sizes, int n_in,
                              void* d_out, int out_size, void* d_ws, size_t ws_size,
                              hipStream_t stream) {
  const float* x   = (const float*)d_in[0];
  const float* W1a = (const float*)d_in[1];
  const float* b1a = (const float*)d_in[2];
  const float* W1b = (const float*)d_in[3];
  const float* b1b = (const float*)d_in[4];
  const float* g1  = (const float*)d_in[5];
  const float* be1 = (const float*)d_in[6];
  const float* W2a = (const float*)d_in[7];
  const float* b2a = (const float*)d_in[8];
  const float* W2b = (const float*)d_in[9];
  const float* b2b = (const float*)d_in[10];
  const float* g2  = (const float*)d_in[11];
  const float* be2 = (const float*)d_in[12];

  const int N = 1024, F = 128, H = 256;

  char* ws = (char*)d_ws;
  size_t off = 0;
  auto alloc = [&](size_t bytes) -> char* {
    char* p = ws + off;
    off = (off + bytes + 255) & ~(size_t)255;
    return p;
  };

  unsigned short* Bt1a = (unsigned short*)alloc((size_t)F * H * 2);   // [H,F]
  unsigned short* Bt1b = (unsigned short*)alloc((size_t)H * H * 2);   // [H,H]
  unsigned short* Bt2a = (unsigned short*)alloc((size_t)H * H * 2);   // [H,H]
  unsigned short* Bt2b = (unsigned short*)alloc((size_t)H * F * 2);   // [F,H]
  float*          S     = (float*)alloc((size_t)H * 4);
  float*          stats = (float*)alloc((size_t)2 * H * 4);
  float*          partb = (float*)alloc((size_t)2 * RB * H * 4);
  unsigned short* aggA  = (unsigned short*)alloc((size_t)N * H * 2);
  unsigned short* Hbf   = (unsigned short*)alloc((size_t)N * H * 2);
  float*          Hf    = (float*)alloc((size_t)N * H * 4);
  float*          BN1   = (float*)alloc((size_t)N * H * 4);
  (void)ws_size; (void)in_sizes; (void)n_in; (void)out_size;

  // ---- weights -> bf16 transposed [Nc,K] (TDM-friendly row-major tiles) ----
  cvt_bf16_t_kernel<<<(F * H + 255) / 256, 256, 0, stream>>>(W1a, Bt1a, F, H);
  cvt_bf16_t_kernel<<<(H * H + 255) / 256, 256, 0, stream>>>(W1b, Bt1b, H, H);
  cvt_bf16_t_kernel<<<(H * H + 255) / 256, 256, 0, stream>>>(W2a, Bt2a, H, H);
  cvt_bf16_t_kernel<<<(H * F + 255) / 256, 256, 0, stream>>>(W2b, Bt2b, H, F);

  // ---- layer 1: agg -> MLP1 -> relu -> BN1 ----
  colsum_part<<<dim3(1, RB), 256, 0, stream>>>(x, partb, N, F);
  colsum_fin<<<1, 256, 0, stream>>>(partb, S, F);
  agg_bf16_kernel<<<(N * F + 255) / 256, 256, 0, stream>>>(x, S, aggA, N, F);
  gemm_bf16_wmma<1, 1><<<dim3(H / TN, N / TM), 256, 0, stream>>>(aggA, Bt1a, b1a, Hbf, N, F, H);
  gemm_bf16_wmma<0, 0><<<dim3(H / TN, N / TM), 256, 0, stream>>>(Hbf, Bt1b, b1b, Hf, N, H, H);
  bnstats_part<<<dim3(1, RB), 256, 0, stream>>>(Hf, partb, N, H);
  bnstats_fin<<<1, 256, 0, stream>>>(partb, stats, N, H);
  bn_apply_kernel<<<(N * H + 255) / 256, 256, 0, stream>>>(Hf, stats, g1, be1, BN1, N, H);

  // ---- layer 2: agg -> MLP2 -> relu -> BN2 -> row-sum readout ----
  colsum_part<<<dim3(1, RB), 256, 0, stream>>>(BN1, partb, N, H);
  colsum_fin<<<1, 256, 0, stream>>>(partb, S, H);
  agg_bf16_kernel<<<(N * H + 255) / 256, 256, 0, stream>>>(BN1, S, aggA, N, H);
  gemm_bf16_wmma<1, 1><<<dim3(H / TN, N / TM), 256, 0, stream>>>(aggA, Bt2a, b2a, Hbf, N, H, H);
  gemm_bf16_wmma<0, 0><<<dim3(F / TN, N / TM), 256, 0, stream>>>(Hbf, Bt2b, b2b, Hf, N, H, F);
  bnstats_part<<<dim3(1, RB), 128, 0, stream>>>(Hf, partb, N, F);
  bnstats_fin<<<1, 128, 0, stream>>>(partb, stats, N, F);
  bnred_part<<<dim3(1, RB), 128, 0, stream>>>(Hf, stats, g2, be2, partb, N, F);
  bnred_fin<<<1, 128, 0, stream>>>(partb, (float*)d_out, F);
}
